// MixtureOfMixers_78795470012703
// MI455X (gfx1250) — matrix operand order repaired and checked by reading, not compiled, not run on hardware
//
#include <hip/hip_runtime.h>
#include <hip/hip_bf16.h>

typedef __attribute__((ext_vector_type(16))) __bf16 v16bf;
typedef __attribute__((ext_vector_type(8)))  float  v8f;

#define BM 128
#define BN 128
#define BK 32
#define LDSW 20  // LDS row stride in 32-bit words (40 bf16 = 80B: 16B-aligned, conflict-free)

#define EPI_H     0
#define EPI_TOKEN 1
#define EPI_CHAN  2

// Hardware packed f32->bf16 (RNE): 2 elements in 1 VALU op.
__device__ __forceinline__ unsigned cvt_pk_bf16(float lo, float hi) {
  unsigned r;
  asm("v_cvt_pk_bf16_f32 %0, %1, %2" : "=v"(r) : "v"(lo), "v"(hi));
  return r;
}

__device__ __forceinline__ unsigned short f2bf(float f) {
  return (unsigned short)cvt_pk_bf16(f, f);
}

// Branch-free tanh-GELU: tanh(u) = 1 - 2/(1+exp(2u)); exp->inf/0 saturates correctly.
__device__ __forceinline__ float gelu_fast(float x) {
  float u = 0.7978845608028654f * (x + 0.044715f * x * x * x);
  float e = __expf(2.0f * u);
  float t = 1.0f - 2.0f * __builtin_amdgcn_rcpf(1.0f + e);
  return 0.5f * x * (1.0f + t);
}

__device__ __forceinline__ void wait_async0() {
  asm volatile("s_wait_asynccnt 0x0" ::: "memory");
}

// Async global->LDS copy of a 128x32 bf16 tile (no VGPR round trip, ASYNCcnt).
__device__ __forceinline__ void async_stage(unsigned* ldsbuf, const unsigned short* src,
                                            int row0, int ld, int k0, int tid) {
#pragma unroll
  for (int j = 0; j < 2; ++j) {
    int idx = tid + j * 256;
    int rr = idx >> 2, q = idx & 3;             // 4 x b128 per 32-elem row
    unsigned voff   = (unsigned)(((size_t)(row0 + rr) * ld + k0) * 2 + q * 16);
    unsigned ldsoff = (unsigned)(size_t)&ldsbuf[rr * LDSW + q * 4];
    asm volatile("global_load_async_to_lds_b128 %0, %1, %2"
                 :: "v"(ldsoff), "v"(voff), "s"(src) : "memory");
  }
}

// f32 operand: prefetch into registers (overlaps WMMA), convert+store later.
__device__ __forceinline__ void load_f32_regs(float4 (&r)[4], const float* s,
                                              int row0, int ld, int k0, int tid) {
#pragma unroll
  for (int j = 0; j < 4; ++j) {
    int idx = tid + j * 256;
    int rr = idx >> 3, q = idx & 7;
    r[j] = *(const float4*)(s + (size_t)(row0 + rr) * ld + k0 + q * 4);
  }
}

__device__ __forceinline__ void store_bf16_lds(unsigned* lds, const float4 (&r)[4], int tid) {
#pragma unroll
  for (int j = 0; j < 4; ++j) {
    int idx = tid + j * 256;
    int rr = idx >> 3, q = idx & 7;
    uint2 p;
    p.x = cvt_pk_bf16(r[j].x, r[j].y);
    p.y = cvt_pk_bf16(r[j].z, r[j].w);
    *(uint2*)&lds[rr * LDSW + q * 2] = p;
  }
}

// Generic NT GEMM: C[m,n] = sum_k A[m,k] * B[n,k]  (both K-contiguous rows)
// EPI_H:     H[m,n]    = bf16(gelu(C + bias[n]))
// EPI_TOKEN: out[b, m(=token), d] += w * (C + bias[m]),  n = b*768 + d
// EPI_CHAN:  out[b, t, n(=d)]     += w * (C + bias[n]),  m = b*256 + t
template<int EPI, bool AF32, bool BF32>
__global__ __launch_bounds__(256)
void gemm_nt(const void* __restrict__ A, const void* __restrict__ Bm,
             int M, int N, int K, int lda, int ldb,
             const float* __restrict__ bias,
             unsigned short* __restrict__ H, int ldh,
             float* __restrict__ out,
             const float* __restrict__ wper, int eid,
             int rows_per_b, int skip_on_cols)
{
  const int tid = threadIdx.x;
  const int bx = blockIdx.x, by = blockIdx.y;

  const int btile = skip_on_cols ? bx : by;
  const int bb = (btile * 128) / rows_per_b;    // batch element of this block
  const float w = wper[bb * 8 + eid];
  if (w == 0.0f) return;                        // MoE top-2 sparsity: skip expert

  __shared__ __align__(16) unsigned ldsA[2][BM * LDSW];
  __shared__ __align__(16) unsigned ldsB[2][BN * LDSW];

  const int lane = tid & 31;
  const int wid  = tid >> 5;
  const int wm = (wid & 3) * 32;   // 4 waves along M
  const int wn = (wid >> 2) * 64;  // 2 waves along N
  const int lm = lane & 15, g = lane >> 4;

  v8f acc[8];
  const v8f zero = {0.f,0.f,0.f,0.f,0.f,0.f,0.f,0.f};
#pragma unroll
  for (int i = 0; i < 8; ++i) acc[i] = zero;

  float4 ra[4], rb[4];

  // ---- prologue: stage k=0 into buffer 0 ----
  if (AF32) { load_f32_regs(ra, (const float*)A, by * BM, lda, 0, tid);
              store_bf16_lds(ldsA[0], ra, tid); }
  else      { async_stage(ldsA[0], (const unsigned short*)A, by * BM, lda, 0, tid); }
  if (BF32) { load_f32_regs(rb, (const float*)Bm, bx * BN, ldb, 0, tid);
              store_bf16_lds(ldsB[0], rb, tid); }
  else      { async_stage(ldsB[0], (const unsigned short*)Bm, bx * BN, ldb, 0, tid); }
  if (!AF32 || !BF32) wait_async0();
  __syncthreads();

  const int nk = K / BK;
  for (int ki = 0; ki < nk; ++ki) {
    const int cur = ki & 1, nxt = cur ^ 1;
    const bool more = (ki + 1) < nk;

    // ---- prefetch next tile: async bf16 -> LDS[nxt], f32 -> regs ----
    if (more) {
      const int k0n = (ki + 1) * BK;
      if (AF32) load_f32_regs(ra, (const float*)A, by * BM, lda, k0n, tid);
      else      async_stage(ldsA[nxt], (const unsigned short*)A, by * BM, lda, k0n, tid);
      if (BF32) load_f32_regs(rb, (const float*)Bm, bx * BN, ldb, k0n, tid);
      else      async_stage(ldsB[nxt], (const unsigned short*)Bm, bx * BN, ldb, k0n, tid);
    }

    // ---- compute on LDS[cur] ----
    union Frag { unsigned u[8]; v16bf v; } fa[2], fb[4];
#pragma unroll
    for (int mt = 0; mt < 2; ++mt) {            // A 16x32: K = g*16 + 2p
      int m = wm + mt * 16 + lm;
      const unsigned* p = &ldsA[cur][m * LDSW + g * 8];
      *(uint2*)&fa[mt].u[0] = *(const uint2*)(p + 0);
      *(uint2*)&fa[mt].u[2] = *(const uint2*)(p + 2);
      *(uint2*)&fa[mt].u[4] = *(const uint2*)(p + 4);
      *(uint2*)&fa[mt].u[6] = *(const uint2*)(p + 6);
    }
#pragma unroll
    for (int nt = 0; nt < 4; ++nt) {            // B 32x16: K = g*16 + 2p
      int n = wn + nt * 16 + lm;
      const unsigned* p = &ldsB[cur][n * LDSW + g * 8];
      *(uint2*)&fb[nt].u[0] = *(const uint2*)(p + 0);
      *(uint2*)&fb[nt].u[2] = *(const uint2*)(p + 2);
      *(uint2*)&fb[nt].u[4] = *(const uint2*)(p + 4);
      *(uint2*)&fb[nt].u[6] = *(const uint2*)(p + 6);
    }
#pragma unroll
    for (int mt = 0; mt < 2; ++mt)
#pragma unroll
      for (int nt = 0; nt < 4; ++nt)
        acc[mt * 4 + nt] = __builtin_amdgcn_wmma_f32_16x16x32_bf16(
            false, fa[mt].v, false, fb[nt].v, (short)0, acc[mt * 4 + nt],
            false, false);

    // ---- commit prefetched f32 operand, close out async, one barrier ----
    if (more) {
      if (AF32) store_bf16_lds(ldsA[nxt], ra, tid);
      if (BF32) store_bf16_lds(ldsB[nxt], rb, tid);
      if (!AF32 || !BF32) wait_async0();
    }
    __syncthreads();
  }

  // ---- epilogue. C layout: lane (g,lm) -> col lm, rows g*8 + vg ----
#pragma unroll
  for (int mt = 0; mt < 2; ++mt) {
#pragma unroll
    for (int nt = 0; nt < 4; ++nt) {
      v8f c = acc[mt * 4 + nt];
      int gm0 = by * BM + wm + mt * 16 + g * 8;
      int gn  = bx * BN + wn + nt * 16 + lm;
#pragma unroll
      for (int vg = 0; vg < 8; ++vg) {
        int gm = gm0 + vg;
        float val = c[vg];
        if (EPI == EPI_H) {
          float h = gelu_fast(val + bias[gn]);
          H[(size_t)gm * ldh + gn] = f2bf(h);
        } else if (EPI == EPI_TOKEN) {
          int d = gn - bb * 768;
          float* o = out + (size_t)bb * 196608 + (size_t)gm * 768 + d;
          *o += w * (val + bias[gm]);
        } else { // EPI_CHAN
          int tkn = gm - bb * 256;
          float* o = out + (size_t)bb * 196608 + (size_t)tkn * 768 + gn;
          *o += w * (val + bias[gn]);
        }
      }
    }
  }
}

__global__ __launch_bounds__(256)
void zero_kernel(float4* __restrict__ p, long n4) {
  long i = (long)blockIdx.x * 256 + threadIdx.x;
  if (i < n4) { float4 z = {0.f, 0.f, 0.f, 0.f}; p[i] = z; }
}

// Token norm over N per (b,d) + transpose to (b*768+d, n) bf16; emits token means.
__global__ __launch_bounds__(256)
void token_norm_kernel(const float* __restrict__ x, unsigned short* __restrict__ xt,
                       float* __restrict__ mean_bd)
{
  __shared__ float tile[256][33];
  __shared__ float ps[8][32], pq[8][32];
  __shared__ float s_mean[32], s_rstd[32];
  const int t = threadIdx.x;
  const int b = blockIdx.x / 24, dt = blockIdx.x % 24;
  const int d0 = dt * 32;
  const float* xb = x + (size_t)b * 196608;
#pragma unroll 4
  for (int j = 0; j < 32; ++j) {
    int idx = j * 256 + t;
    int n = idx >> 5, dj = idx & 31;
    tile[n][dj] = xb[(size_t)n * 768 + d0 + dj];
  }
  __syncthreads();
  {
    int dj = t & 31, seg = t >> 5;
    float s = 0.f, q = 0.f;
    for (int i = 0; i < 32; ++i) {
      float v = tile[seg * 32 + i][dj];
      s += v; q += v * v;
    }
    ps[seg][dj] = s; pq[seg][dj] = q;
  }
  __syncthreads();
  if (t < 32) {
    float s = 0.f, q = 0.f;
    for (int seg = 0; seg < 8; ++seg) { s += ps[seg][t]; q += pq[seg][t]; }
    float mean = s * (1.0f / 256.0f);
    float var  = q * (1.0f / 256.0f) - mean * mean;
    s_mean[t] = mean;
    s_rstd[t] = rsqrtf(var + 1e-6f);
    mean_bd[(size_t)b * 768 + d0 + t] = mean;
  }
  __syncthreads();
#pragma unroll 4
  for (int j = 0; j < 32; ++j) {
    float v = (tile[t][j] - s_mean[j]) * s_rstd[j];
    xt[((size_t)b * 768 + d0 + j) * 256 + t] = f2bf(v);
  }
}

// Channel norm over D per (b,n) -> bf16 (b*256+n, d)
__global__ __launch_bounds__(256)
void chan_norm_kernel(const float* __restrict__ x, unsigned short* __restrict__ xn)
{
  __shared__ float red[256], red2[256];
  __shared__ float s_mean, s_rstd;
  const int t = threadIdx.x;
  const size_t base = (size_t)blockIdx.x * 768;
  float v0 = x[base + t], v1 = x[base + 256 + t], v2 = x[base + 512 + t];
  red[t]  = v0 + v1 + v2;
  red2[t] = v0 * v0 + v1 * v1 + v2 * v2;
  __syncthreads();
  for (int off = 128; off > 0; off >>= 1) {
    if (t < off) { red[t] += red[t + off]; red2[t] += red2[t + off]; }
    __syncthreads();
  }
  if (t == 0) {
    float mean = red[0] * (1.f / 768.f);
    float var  = red2[0] * (1.f / 768.f) - mean * mean;
    s_mean = mean;
    s_rstd = rsqrtf(var + 1e-6f);
  }
  __syncthreads();
  float mean = s_mean, rstd = s_rstd;
  xn[base + t]       = f2bf((v0 - mean) * rstd);
  xn[base + 256 + t] = f2bf((v1 - mean) * rstd);
  xn[base + 512 + t] = f2bf((v2 - mean) * rstd);
}

// Router: logits -> softmax -> top-2 renorm -> w_per; aux loss scalar.
__global__ __launch_bounds__(256)
void router_kernel(const float* __restrict__ mean_bd, const float* __restrict__ router_w,
                   float* __restrict__ wper, float* __restrict__ aux_out)
{
  __shared__ float logits[32][8];
  __shared__ float probs[32][8];
  __shared__ int   top1[32];
  const int t = threadIdx.x;
  {
    int b = t >> 3, e = t & 7;
    const float* mb = mean_bd + (size_t)b * 768;
    const float* rw = router_w + (size_t)e * 768;
    float s = 0.f;
    for (int d = 0; d < 768; ++d) s += mb[d] * rw[d];
    logits[b][e] = s;
  }
  __syncthreads();
  if (t < 32) {
    int b = t;
    float mx = logits[b][0];
    for (int e = 1; e < 8; ++e) mx = fmaxf(mx, logits[b][e]);
    float p[8], sum = 0.f;
    for (int e = 0; e < 8; ++e) { p[e] = expf(logits[b][e] - mx); sum += p[e]; }
    for (int e = 0; e < 8; ++e) { p[e] /= sum; probs[b][e] = p[e]; }
    int i1 = 0;
    for (int e = 1; e < 8; ++e) if (p[e] > p[i1]) i1 = e;
    int i2 = (i1 == 0) ? 1 : 0;
    for (int e = 0; e < 8; ++e) if (e != i1 && p[e] > p[i2]) i2 = e;
    float wsum = p[i1] + p[i2];
    for (int e = 0; e < 8; ++e) wper[b * 8 + e] = 0.f;
    wper[b * 8 + i1] = p[i1] / wsum;
    wper[b * 8 + i2] = p[i2] / wsum;
    top1[b] = i1;
  }
  __syncthreads();
  if (t == 0) {
    float aux = 0.f;
    for (int e = 0; e < 8; ++e) {
      float pm = 0.f, fr = 0.f;
      for (int b = 0; b < 32; ++b) {
        pm += probs[b][e];
        fr += (top1[b] == e) ? 1.f : 0.f;
      }
      aux += (pm * (1.f / 32.f)) * (fr * (1.f / 32.f));
    }
    aux_out[0] = 8.f * aux;
  }
}

extern "C" void kernel_launch(void* const* d_in, const int* in_sizes, int n_in,
                              void* d_out, int out_size, void* d_ws, size_t ws_size,
                              hipStream_t stream)
{
  const float* x        = (const float*)d_in[0];
  const float* router_w = (const float*)d_in[1];
  const float* tw1 = (const float*)d_in[2];
  const float* tb1 = (const float*)d_in[3];
  const float* tw2 = (const float*)d_in[4];
  const float* tb2 = (const float*)d_in[5];
  const float* cw1 = (const float*)d_in[6];
  const float* cb1 = (const float*)d_in[7];
  const float* cw2 = (const float*)d_in[8];
  const float* cb2 = (const float*)d_in[9];

  float* out = (float*)d_out;       // (32,256,768) f32
  float* aux = out + 6291456;       // scalar aux loss

  char* ws = (char*)d_ws;
  float*          wper    = (float*)(ws + 0);          //   256 f32
  float*          mean_bd = (float*)(ws + 1024);       // 24576 f32
  unsigned short* xt      = (unsigned short*)(ws + 99328);     // 24576x256 bf16
  unsigned short* xn      = (unsigned short*)(ws + 12682240);  // 8192x768 bf16
  unsigned short* Hb      = (unsigned short*)(ws + 25265152);  // 25.2M bf16 (shared h)

  zero_kernel<<<6144, 256, 0, stream>>>((float4*)out, 1572864L);
  token_norm_kernel<<<768, 256, 0, stream>>>(x, xt, mean_bd);
  chan_norm_kernel<<<8192, 256, 0, stream>>>(x, xn);
  router_kernel<<<1, 256, 0, stream>>>(mean_bd, router_w, wper, aux);

  for (int e = 0; e < 4; ++e) {  // token-mixer experts
    gemm_nt<EPI_H, false, true><<<dim3(8, 192), 256, 0, stream>>>(
        xt, tw1 + (size_t)e * 1024 * 256, 24576, 1024, 256, 256, 256,
        tb1 + e * 1024, Hb, 1024, nullptr, wper, e, 768, 0);
    // transposed second GEMM: C'[token, b*768+d] -> coalesced out writes
    gemm_nt<EPI_TOKEN, true, false><<<dim3(192, 2), 256, 0, stream>>>(
        tw2 + (size_t)e * 256 * 1024, Hb, 256, 24576, 1024, 1024, 1024,
        tb2 + e * 256, nullptr, 0, out, wper, e, 768, 1);
  }
  for (int e = 0; e < 4; ++e) {  // channel-mixer experts
    gemm_nt<EPI_H, false, true><<<dim3(24, 64), 256, 0, stream>>>(
        xn, cw1 + (size_t)e * 3072 * 768, 8192, 3072, 768, 768, 768,
        cb1 + e * 3072, Hb, 3072, nullptr, wper, 4 + e, 256, 0);
    gemm_nt<EPI_CHAN, false, true><<<dim3(6, 64), 256, 0, stream>>>(
        Hb, cw2 + (size_t)e * 768 * 3072, 8192, 768, 3072, 3072, 3072,
        cb2 + e * 768, nullptr, 0, out, wper, 4 + e, 256, 0);
  }
}
